// VisionTransformer_90125593739474
// MI455X (gfx1250) — matrix operand order repaired and checked by reading, compiled, not run on hardware
//
#include <hip/hip_runtime.h>
#include <hip/hip_bf16.h>
#include <math.h>

typedef unsigned short u16;
typedef __attribute__((ext_vector_type(16))) __bf16 v16bf;
typedef __attribute__((ext_vector_type(8)))  float  v8f;
typedef __attribute__((ext_vector_type(4))) unsigned int u32x4;
typedef __attribute__((ext_vector_type(8))) int i32x8;
typedef __attribute__((ext_vector_type(4))) int i32x4;

#define DEPTH 12
#define EDIM  768
#define NHEAD 12
#define HID   3072
#define TSEQ  577
#define BATCH 16
#define NCLS  1000

#if defined(__has_builtin)
#if __has_builtin(__builtin_amdgcn_tensor_load_to_lds) && __has_builtin(__builtin_amdgcn_s_wait_tensorcnt)
#define HAVE_TDM 1
#endif
#endif

// ---------------------------------------------------------------- helpers
__device__ __forceinline__ u16 f2bf(float f) {
    union { float f; unsigned u; } c; c.f = f;
    unsigned u = c.u;
    unsigned r = u + 0x7fffu + ((u >> 16) & 1u);
    if ((u & 0x7fffffffu) > 0x7f800000u) r = u | 0x00400000u;  // NaN
    return (u16)(r >> 16);
}
__device__ __forceinline__ float bf2f(u16 h) {
    union { unsigned u; float f; } c; c.u = ((unsigned)h) << 16; return c.f;
}

struct Frag { union { v16bf v; uint4 q[2]; u16 u[16]; }; };

// A fragment: 16x32 bf16, M x K. lane&15 = row; lane<16 holds K {k0..k0+7, k0+16..k0+23},
// lane>=16 holds K {k0+8..k0+15, k0+24..k0+31}  (CDNA5 16-bit A layout)
__device__ __forceinline__ v16bf load_a(const u16* __restrict__ base, int lda, int row,
                                        int k0, int lane) {
    const u16* p = base + (size_t)row * lda + k0 + ((lane & 16) ? 8 : 0);
    Frag f;
    f.q[0] = *(const uint4*)(p);
    f.q[1] = *(const uint4*)(p + 16);
    return f.v;
}
// B fragment: 32x16 bf16, K x N, sourced from row-major W[N][K] (so B = W^T).
// lane&15 = column n; lane<16 holds K {k0..k0+7, k0+16..k0+23}, lane>=16 holds {k0+8..k0+15, k0+24..k0+31}.
__device__ __forceinline__ v16bf load_b(const u16* __restrict__ base, int ldb, int col,
                                        int k0, int lane) {
    const u16* p = base + (size_t)col * ldb + k0 + ((lane & 16) ? 8 : 0);
    Frag f;
    f.q[0] = *(const uint4*)(p);
    f.q[1] = *(const uint4*)(p + 16);
    return f.v;
}
__device__ __forceinline__ v8f wmma_bf16(v16bf a, v16bf b, v8f c) {
    return __builtin_amdgcn_wmma_f32_16x16x32_bf16(false, a, false, b, (short)0, c,
                                                   false, false);
}

// ---------------------------------------------------------------- f32 -> bf16 convert
__global__ void k_f2bf(const float* __restrict__ src, u16* __restrict__ dst, int n) {
    for (int i = blockIdx.x * blockDim.x + threadIdx.x; i < n; i += gridDim.x * blockDim.x)
        dst[i] = f2bf(src[i]);
}

// ---------------------------------------------------------------- patchify (to bf16)
// xp[b*576 + gy*24+gx][ch*256 + py*16 + px] = x[b][ch][gy*16+py][gx*16+px]
__global__ void k_patchify(const float* __restrict__ x, u16* __restrict__ xp, int total) {
    for (int idx = blockIdx.x * blockDim.x + threadIdx.x; idx < total;
         idx += gridDim.x * blockDim.x) {
        int c   = idx % EDIM;
        int row = idx / EDIM;
        int b   = row / 576;
        int p   = row % 576;
        int gy  = p / 24, gx = p % 24;
        int ch  = c / 256;
        int rem = c % 256;
        int py  = rem / 16, px = rem % 16;
        size_t src = (((size_t)(b * 3 + ch) * 384) + gy * 16 + py) * 384 + gx * 16 + px;
        xp[idx] = f2bf(x[src]);
    }
}

// ---------------------------------------------------------------- cls/pos assembly
__global__ void k_assemble(const float* __restrict__ pout, const float* __restrict__ cls,
                           const float* __restrict__ pos, float* __restrict__ h, int total) {
    for (int idx = blockIdx.x * blockDim.x + threadIdx.x; idx < total;
         idx += gridDim.x * blockDim.x) {
        int e  = idx % EDIM;
        int bt = idx / EDIM;
        int t  = bt % TSEQ;
        int b  = bt / TSEQ;
        float v = (t == 0) ? cls[e] : pout[((size_t)(b * 576 + t - 1)) * EDIM + e];
        h[idx] = v + pos[(size_t)t * EDIM + e];
    }
}

// ---------------------------------------------------------------- LayerNorm (E=768)
// one block per output row; input row index = blockIdx.x * row_stride
__global__ __launch_bounds__(256)
void k_ln(const float* __restrict__ x, const float* __restrict__ g,
          const float* __restrict__ bb, u16* __restrict__ outH,
          float* __restrict__ outF, int row_stride) {
    __shared__ float red[256];
    const int tid = threadIdx.x;
    const float* xr = x + (size_t)blockIdx.x * row_stride * EDIM;
    float v0 = xr[tid], v1 = xr[tid + 256], v2 = xr[tid + 512];
    red[tid] = v0 + v1 + v2;
    __syncthreads();
    for (int off = 128; off > 0; off >>= 1) {
        if (tid < off) red[tid] += red[tid + off];
        __syncthreads();
    }
    float mean = red[0] * (1.0f / EDIM);
    __syncthreads();
    float d0 = v0 - mean, d1 = v1 - mean, d2 = v2 - mean;
    red[tid] = d0 * d0 + d1 * d1 + d2 * d2;
    __syncthreads();
    for (int off = 128; off > 0; off >>= 1) {
        if (tid < off) red[tid] += red[tid + off];
        __syncthreads();
    }
    float inv = rsqrtf(red[0] * (1.0f / EDIM) + 1e-6f);
    size_t o = (size_t)blockIdx.x * EDIM;
    float y0 = d0 * inv * g[tid]       + bb[tid];
    float y1 = d1 * inv * g[tid + 256] + bb[tid + 256];
    float y2 = d2 * inv * g[tid + 512] + bb[tid + 512];
    if (outH) { outH[o + tid] = f2bf(y0); outH[o + tid + 256] = f2bf(y1); outH[o + tid + 512] = f2bf(y2); }
    if (outF) { outF[o + tid] = y0;       outF[o + tid + 256] = y1;       outF[o + tid + 512] = y2; }
}

// ---------------------------------------------------------------- generic bf16 WMMA GEMM
// C[M,N] = A[M,K] (bf16, row-major) @ W[N,K]^T (bf16) + bias, optional GELU, optional
// f32 residual add; writes f32 and/or bf16.
// Wave -> 32x64 strip (2 M-tiles x 4 N-tiles, B fragments reused), block(256) -> 256x64.
__global__ __launch_bounds__(256)
void k_gemm(const u16* __restrict__ A, const u16* __restrict__ W,
            const float* __restrict__ bias, int M, int N, int K,
            float* __restrict__ outF, u16* __restrict__ outH,
            const float* __restrict__ resid, int gelu_flag) {
    const int lane = threadIdx.x & 31;
    const int wave = threadIdx.x >> 5;
    const int m0 = (blockIdx.y * 8 + wave) * 32;
    const int n0 = blockIdx.x * 64;
    if (m0 >= M) return;  // wave-uniform: EXEC stays all-ones for WMMA

    v8f acc[2][4];
#pragma unroll
    for (int t = 0; t < 2; ++t)
#pragma unroll
        for (int j = 0; j < 4; ++j)
#pragma unroll
            for (int r = 0; r < 8; ++r) acc[t][j][r] = 0.0f;

    int r0 = m0 + (lane & 15);            // always < M (M is a multiple of 16)
    int r1 = m0 + 16 + (lane & 15);
    int r1c = (r1 < M) ? r1 : (M - 1);    // clamp loads for the tail wave
    for (int k0 = 0; k0 < K; k0 += 32) {
        v16bf a0 = load_a(A, K, r0,  k0, lane);
        v16bf a1 = load_a(A, K, r1c, k0, lane);
#pragma unroll
        for (int j = 0; j < 4; ++j) {
            v16bf b = load_b(W, K, n0 + 16 * j + (lane & 15), k0, lane);
            acc[0][j] = wmma_bf16(a0, b, acc[0][j]);
            acc[1][j] = wmma_bf16(a1, b, acc[1][j]);
        }
    }

#pragma unroll
    for (int t = 0; t < 2; ++t) {
        const int mbase = m0 + 16 * t + ((lane & 16) ? 8 : 0);
#pragma unroll
        for (int j = 0; j < 4; ++j) {
            int col = n0 + 16 * j + (lane & 15);
            float bv = bias ? bias[col] : 0.0f;
#pragma unroll
            for (int r = 0; r < 8; ++r) {
                int row = mbase + r;
                if (row < M) {
                    float v = acc[t][j][r] + bv;
                    if (gelu_flag) v = 0.5f * v * (1.0f + erff(v * 0.70710678118654752f));
                    size_t o = (size_t)row * N + col;
                    if (resid) v += resid[o];
                    if (outF) outF[o] = v;
                    if (outH) outH[o] = f2bf(v);
                }
            }
        }
    }
}

// ---------------------------------------------------------------- flash attention
// one wave per (b,h, 16-query tile). qkv packed [B*T, 3E]; ctx bf16 [B*T, E].
// V tiles staged into LDS by the Tensor Data Mover (TDM) when available.
__global__ __launch_bounds__(32)
void k_attn(const u16* __restrict__ qkv, u16* __restrict__ ctx) {
    const int lane = threadIdx.x;
    const int b = blockIdx.x / NHEAD;
    const int h = blockIdx.x % NHEAD;
    const int q0 = blockIdx.y * 16;
    const float scale = 0.125f;  // 64^-0.5

    const u16* Qb = qkv + (size_t)b * TSEQ * (3 * EDIM) + h * 64;
    const u16* Kb = Qb + EDIM;
    const u16* Vb = Qb + 2 * EDIM;

    __shared__ __align__(16) float s_S[16][32];
    __shared__ __align__(16) u16   s_P[16][32];
    __shared__ __align__(16) u16   s_V[32][64];
    __shared__ float s_stat[3][16];  // 0:max 1:sum 2:alpha/scale

    if (lane < 16) { s_stat[0][lane] = -1e30f; s_stat[1][lane] = 0.0f; }

    int qrow = q0 + (lane & 15); if (qrow > TSEQ - 1) qrow = TSEQ - 1;
    v16bf qa0 = load_a(Qb, 3 * EDIM, qrow, 0, lane);
    v16bf qa1 = load_a(Qb, 3 * EDIM, qrow, 32, lane);

    v8f acc[4];
#pragma unroll
    for (int j = 0; j < 4; ++j)
#pragma unroll
        for (int r = 0; r < 8; ++r) acc[j][r] = 0.0f;

    for (int kt = 0; kt < TSEQ; kt += 32) {
#ifdef HAVE_TDM
        // ---- TDM: DMA the 32x64 V tile (row stride 3E) into LDS; rows past T
        //      are out-of-tensor and auto-zero-filled by the TDM.
        {
            unsigned lds_off = (unsigned)(size_t)(&s_V[0][0]);
            unsigned long long gaddr =
                (unsigned long long)(const void*)(Vb + (size_t)kt * (3 * EDIM));
            unsigned rows = (unsigned)(TSEQ - kt);  // tensor_dim1 (OOB -> zero)
            u32x4 g0;
            g0[0] = 1u;                                   // count=1 (valid user D#)
            g0[1] = lds_off;                              // lds_addr
            g0[2] = (unsigned)(gaddr & 0xffffffffu);      // global_addr[31:0]
            g0[3] = (unsigned)((gaddr >> 32) & 0x01ffffffu) | (2u << 30);  // type=2
            i32x8 g1;
            g1[0] = (int)(1u << 16);                      // data_size = 2 bytes
            g1[1] = (int)(64u << 16);                     // tensor_dim0 = 64 (lo16)
            g1[2] = (int)((rows & 0xffffu) << 16);        // dim0 hi=0 | tensor_dim1 lo
            g1[3] = (int)(64u << 16);                     // dim1 hi=0 | tile_dim0 = 64
            g1[4] = 32;                                   // tile_dim1=32, tile_dim2=0
            g1[5] = 3 * EDIM;                             // tensor_dim0_stride = 2304
            g1[6] = 0;
            g1[7] = 0;
            i32x4 gz; gz[0] = gz[1] = gz[2] = gz[3] = 0;  // groups 2/3 unused (2D)
#if __clang_major__ >= 23
            i32x8 gz8;
#pragma unroll
            for (int i = 0; i < 8; ++i) gz8[i] = 0;
            __builtin_amdgcn_tensor_load_to_lds(g0, g1, gz, gz, gz8, 0);
#else
            __builtin_amdgcn_tensor_load_to_lds(g0, g1, gz, gz, 0);
#endif
        }
#else
        // ---- fallback: cooperative V tile load, lane -> one key row
        {
            int vr = kt + lane; if (vr > TSEQ - 1) vr = TSEQ - 1;
            const uint4* vsrc = (const uint4*)(Vb + (size_t)vr * (3 * EDIM));
            uint4* vdst = (uint4*)(&s_V[lane][0]);
#pragma unroll
            for (int i = 0; i < 8; ++i) vdst[i] = vsrc[i];
        }
#endif
        // scores for two 16-key tiles (overlaps with the V DMA)
#pragma unroll
        for (int jt = 0; jt < 2; ++jt) {
            int krow = kt + 16 * jt + (lane & 15);
            if (krow > TSEQ - 1) krow = TSEQ - 1;
            v16bf kf0 = load_b(Kb, 3 * EDIM, krow, 0, lane);
            v16bf kf1 = load_b(Kb, 3 * EDIM, krow, 32, lane);
            v8f c;
#pragma unroll
            for (int r = 0; r < 8; ++r) c[r] = 0.0f;
            c = wmma_bf16(qa0, kf0, c);
            c = wmma_bf16(qa1, kf1, c);
#pragma unroll
            for (int r = 0; r < 8; ++r)
                s_S[r + ((lane & 16) ? 8 : 0)][16 * jt + (lane & 15)] = c[r] * scale;
        }
        __syncthreads();
        // online softmax update, lanes 0..15 each own one query row
        if (lane < 16) {
            if (q0 + lane < TSEQ) {
                float mold = s_stat[0][lane];
                float mx = mold;
                float sv[32];
#pragma unroll
                for (int i = 0; i < 32; ++i) {
                    float s = (kt + i < TSEQ) ? s_S[lane][i] : -1e30f;
                    sv[i] = s;
                    mx = fmaxf(mx, s);
                }
                float alpha = __expf(mold - mx);
                float sum = s_stat[1][lane] * alpha;
#pragma unroll
                for (int i = 0; i < 32; ++i) {
                    float p = (kt + i < TSEQ) ? __expf(sv[i] - mx) : 0.0f;
                    s_P[lane][i] = f2bf(p);
                    sum += p;
                }
                s_stat[0][lane] = mx;
                s_stat[1][lane] = sum;
                s_stat[2][lane] = alpha;
            } else {
#pragma unroll
                for (int i = 0; i < 32; ++i) s_P[lane][i] = 0;
                s_stat[2][lane] = 0.0f;
                s_stat[1][lane] = 1.0f;
            }
        }
#ifdef HAVE_TDM
        __builtin_amdgcn_s_wait_tensorcnt(0);  // V tile resident in LDS
#endif
        __syncthreads();
        // P fragment (A layout) from LDS
        Frag pf;
        {
            int pr = lane & 15;
            int kb = (lane & 16) ? 8 : 0;
#pragma unroll
            for (int i = 0; i < 8; ++i) {
                pf.u[i]     = s_P[pr][kb + i];
                pf.u[8 + i] = s_P[pr][kb + 16 + i];
            }
        }
        // rescale accumulators + P@V
#pragma unroll
        for (int j = 0; j < 4; ++j) {
            Frag vf;
            int d  = 16 * j + (lane & 15);
            int kb = (lane & 16) ? 8 : 0;
#pragma unroll
            for (int i = 0; i < 16; ++i) vf.u[i] = s_V[(i < 8) ? (kb + i) : (kb + 8 + i)][d];
#pragma unroll
            for (int r = 0; r < 8; ++r) acc[j][r] *= s_stat[2][r + ((lane & 16) ? 8 : 0)];
            acc[j] = wmma_bf16(pf.v, vf.v, acc[j]);
        }
        __syncthreads();
    }
    if (lane < 16) {
        float sum = s_stat[1][lane];
        s_stat[2][lane] = (q0 + lane < TSEQ && sum > 0.0f) ? 1.0f / sum : 0.0f;
    }
    __syncthreads();
#pragma unroll
    for (int j = 0; j < 4; ++j) {
        int d = 16 * j + (lane & 15);
#pragma unroll
        for (int r = 0; r < 8; ++r) {
            int m = r + ((lane & 16) ? 8 : 0);
            int t = q0 + m;
            if (t < TSEQ) {
                float v = acc[j][r] * s_stat[2][m];
                ctx[((size_t)(b * TSEQ + t)) * EDIM + h * 64 + d] = f2bf(v);
            }
        }
    }
}

// ---------------------------------------------------------------- head GEMV (16x1000)
__global__ __launch_bounds__(256)
void k_head(const float* __restrict__ xf, const float* __restrict__ w,
            const float* __restrict__ bias, float* __restrict__ out) {
    int idx = blockIdx.x * blockDim.x + threadIdx.x;
    if (idx >= BATCH * NCLS) return;
    int n = idx % NCLS, b = idx / NCLS;
    const float* xr = xf + (size_t)b * EDIM;
    const float* wr = w + (size_t)n * EDIM;
    float s = bias[n];
    for (int k = 0; k < EDIM; ++k) s += xr[k] * wr[k];
    out[idx] = s;
}

// ---------------------------------------------------------------- launch
extern "C" void kernel_launch(void* const* d_in, const int* in_sizes, int n_in,
                              void* d_out, int out_size, void* d_ws, size_t ws_size,
                              hipStream_t stream) {
    (void)in_sizes; (void)n_in; (void)out_size; (void)ws_size;
    const float* in_x      = (const float*)d_in[0];
    const float* patch_w   = (const float*)d_in[1];
    const float* patch_b   = (const float*)d_in[2];
    const float* cls_token = (const float*)d_in[3];
    const float* pos_embed = (const float*)d_in[4];
    const float* ln1_s     = (const float*)d_in[5];
    const float* ln1_b     = (const float*)d_in[6];
    const float* qkv_w     = (const float*)d_in[7];
    const float* qkv_b     = (const float*)d_in[8];
    const float* proj_w    = (const float*)d_in[9];
    const float* proj_b    = (const float*)d_in[10];
    const float* ln2_s     = (const float*)d_in[11];
    const float* ln2_b     = (const float*)d_in[12];
    const float* fc1_w     = (const float*)d_in[13];
    const float* fc1_b     = (const float*)d_in[14];
    const float* fc2_w     = (const float*)d_in[15];
    const float* fc2_b     = (const float*)d_in[16];
    const float* lnf_s     = (const float*)d_in[17];
    const float* lnf_b     = (const float*)d_in[18];
    const float* head_w    = (const float*)d_in[19];
    const float* head_b    = (const float*)d_in[20];

    const int M  = BATCH * TSEQ;        // 9232
    const int MP = BATCH * 576;         // 9216

    char* ws = (char*)d_ws;
    size_t off = 0;
    auto alloc = [&](size_t bytes) -> char* {
        char* p = ws + off;
        off = (off + bytes + 255) & ~(size_t)255;
        return p;
    };
    u16*   wqkv   = (u16*)alloc((size_t)DEPTH * 3 * EDIM * EDIM * 2);
    u16*   wproj  = (u16*)alloc((size_t)DEPTH * EDIM * EDIM * 2);
    u16*   wfc1   = (u16*)alloc((size_t)DEPTH * HID * EDIM * 2);
    u16*   wfc2   = (u16*)alloc((size_t)DEPTH * EDIM * HID * 2);
    u16*   wpatch = (u16*)alloc((size_t)EDIM * EDIM * 2);
    float* hbuf   = (float*)alloc((size_t)M * EDIM * 4);
    u16*   hn     = (u16*)alloc((size_t)M * EDIM * 2);
    u16*   qkvbuf = (u16*)alloc((size_t)M * 3 * EDIM * 2);
    u16*   ctx    = (u16*)alloc((size_t)M * EDIM * 2);
    u16*   mlp    = (u16*)alloc((size_t)M * HID * 2);
    // aliases (lifetimes disjoint):
    float* patch_out = (float*)qkvbuf;   // used before first qkv GEMM
    u16*   xp        = mlp;              // used before first fc1 GEMM
    float* lnf_f32   = (float*)mlp;      // used after last fc2 GEMM

    // weight conversion fp32 -> bf16
    k_f2bf<<<4096, 256, 0, stream>>>(qkv_w,  wqkv,   DEPTH * 3 * EDIM * EDIM);
    k_f2bf<<<4096, 256, 0, stream>>>(proj_w, wproj,  DEPTH * EDIM * EDIM);
    k_f2bf<<<4096, 256, 0, stream>>>(fc1_w,  wfc1,   DEPTH * HID * EDIM);
    k_f2bf<<<4096, 256, 0, stream>>>(fc2_w,  wfc2,   DEPTH * EDIM * HID);
    k_f2bf<<<2048, 256, 0, stream>>>(patch_w, wpatch, EDIM * EDIM);

    // patch embedding
    k_patchify<<<4096, 256, 0, stream>>>(in_x, xp, MP * EDIM);
    k_gemm<<<dim3(EDIM / 64, (MP + 255) / 256), 256, 0, stream>>>(
        xp, wpatch, patch_b, MP, EDIM, EDIM, patch_out, nullptr, nullptr, 0);
    k_assemble<<<4096, 256, 0, stream>>>(patch_out, cls_token, pos_embed, hbuf, M * EDIM);

    for (int i = 0; i < DEPTH; ++i) {
        k_ln<<<M, 256, 0, stream>>>(hbuf, ln1_s + i * EDIM, ln1_b + i * EDIM,
                                    hn, nullptr, 1);
        k_gemm<<<dim3(3 * EDIM / 64, (M + 255) / 256), 256, 0, stream>>>(
            hn, wqkv + (size_t)i * 3 * EDIM * EDIM, qkv_b + i * 3 * EDIM,
            M, 3 * EDIM, EDIM, nullptr, qkvbuf, nullptr, 0);
        k_attn<<<dim3(BATCH * NHEAD, (TSEQ + 15) / 16), 32, 0, stream>>>(qkvbuf, ctx);
        k_gemm<<<dim3(EDIM / 64, (M + 255) / 256), 256, 0, stream>>>(
            ctx, wproj + (size_t)i * EDIM * EDIM, proj_b + i * EDIM,
            M, EDIM, EDIM, hbuf, nullptr, hbuf, 0);
        k_ln<<<M, 256, 0, stream>>>(hbuf, ln2_s + i * EDIM, ln2_b + i * EDIM,
                                    hn, nullptr, 1);
        k_gemm<<<dim3(HID / 64, (M + 255) / 256), 256, 0, stream>>>(
            hn, wfc1 + (size_t)i * HID * EDIM, fc1_b + i * HID,
            M, HID, EDIM, nullptr, mlp, nullptr, 1);
        k_gemm<<<dim3(EDIM / 64, (M + 255) / 256), 256, 0, stream>>>(
            mlp, wfc2 + (size_t)i * EDIM * HID, fc2_b + i * EDIM,
            M, EDIM, HID, hbuf, nullptr, hbuf, 0);
    }

    // final LN on cls tokens only (row stride = TSEQ), then head
    k_ln<<<BATCH, 256, 0, stream>>>(hbuf, lnf_s, lnf_b, nullptr, lnf_f32, TSEQ);
    k_head<<<(BATCH * NCLS + 255) / 256, 256, 0, stream>>>(lnf_f32, head_w, head_b,
                                                           (float*)d_out);
}
